// CtmpGIN_14877766713711
// MI455X (gfx1250) — matrix operand order repaired and checked by reading, not compiled, not run on hardware
//
#include <hip/hip_runtime.h>
#include <hip/hip_bf16.h>
#include <math.h>

#define B_    4096
#define NN_   64
#define C_    128
#define V_    100
#define ED_   64
#define LD_   8
#define E_    4000000
#define N_    (2*B_*NN_)     /* 524288 */
#define HALF_ (B_*NN_)       /* 262144 */

typedef __attribute__((ext_vector_type(16))) _Float16 v16h;
typedef __attribute__((ext_vector_type(8)))  float    v8f;

// ---------------- WMMA fragment helpers (16x16x32 f16 -> f32) ----------------
// A (16x32 f16): lane L holds row M = L%16; halves e=0..7 -> K = kb + (L<16?0:8) + e,
//                e=8..15 -> K = kb + 16 + (L<16?0:8) + (e-8).
__device__ __forceinline__ v16h load_a_frag(const _Float16* row, int kbase, int khalf) {
  v16h a;
#pragma unroll
  for (int e = 0; e < 8; ++e) {
    a[e]     = row[kbase + khalf + e];
    a[e + 8] = row[kbase + 16 + khalf + e];
  }
  return a;
}
// B (32x16 f16) from 64x64 row-major [k][n]: lane L holds column N = ntile*16 + L%16,
// halves e=0..7 -> K = ktile*32 + (L<16?0:8) + e, e=8..15 -> K = ktile*32 + 16 + (L<16?0:8) + (e-8).
__device__ __forceinline__ v16h load_b_frag(const _Float16* W, int ktile, int ntile, int lane) {
  const _Float16* p = W + (ktile * 32 + ((lane >> 4) << 3)) * 64 + ntile * 16 + (lane & 15);
  v16h b;
#pragma unroll
  for (int e = 0; e < 8; ++e) {
    b[e]     = p[e * 64];
    b[e + 8] = p[(16 + e) * 64];
  }
  return b;
}

// ---------------- utility: zero the aggregation buffer ----------------
__global__ __launch_bounds__(256) void zero_kernel(float4* __restrict__ p, unsigned int n4) {
  unsigned int i = blockIdx.x * 256u + threadIdx.x;
  if (i < n4) p[i] = make_float4(0.f, 0.f, 0.f, 0.f);
}

// ---------------- embedding gather: h[n] = col_emb[col(n), x[b(n), col(n)]] ----------------
__global__ __launch_bounds__(256) void embed_kernel(const int* __restrict__ x,
                                                    const float* __restrict__ col_emb,
                                                    float* __restrict__ h) {
  unsigned int t = blockIdx.x * 256u + threadIdx.x;   // [0, N_*32)
  unsigned int n = t >> 5;
  int c2 = (int)(t & 31u) * 2;
  int b, col;
  if (n < (unsigned)HALF_) { b = (int)(n / NN_); col = (int)(n % NN_); }
  else { unsigned int m2 = n - HALF_; b = (int)(m2 / NN_); col = NN_ + (int)(m2 % NN_); }
  int tok = x[(size_t)b * C_ + col];
  const float2 v = *(const float2*)(col_emb + ((size_t)col * V_ + tok) * ED_ + c2);
  *(float2*)(h + (size_t)n * 64 + c2) = v;
}

// ---------------- stage-1 scatter: agg[dst] += h[src] (wave per edge) ----------------
__global__ __launch_bounds__(256) void scatter1_kernel(const float* __restrict__ h,
                                                       const int* __restrict__ src,
                                                       const int* __restrict__ dst,
                                                       float* __restrict__ agg) {
  unsigned int gw = (blockIdx.x * 256u + threadIdx.x) >> 5;
  int lane = threadIdx.x & 31;
  if (gw >= (unsigned)E_) return;
  size_t s = (size_t)src[gw], d = (size_t)dst[gw];
  const float2 v = *(const float2*)(h + s * 64 + lane * 2);
  atomicAdd(agg + d * 64 + lane * 2 + 0, v.x);
  atomicAdd(agg + d * 64 + lane * 2 + 1, v.y);
}

// ---------------- stage-2 interior scatter: agg[dst] += relu(h[src] + base0) ----------------
__global__ __launch_bounds__(256) void scatter2_kernel(const float* __restrict__ h,
                                                       const int* __restrict__ src,
                                                       const int* __restrict__ dst,
                                                       const float* __restrict__ base0,
                                                       float* __restrict__ agg) {
  unsigned int gw = (blockIdx.x * 256u + threadIdx.x) >> 5;
  int lane = threadIdx.x & 31;
  if (gw >= (unsigned)E_) return;
  size_t s = (size_t)src[gw], d = (size_t)dst[gw];
  const float2 v  = *(const float2*)(h + s * 64 + lane * 2);
  const float2 bb = *(const float2*)(base0 + lane * 2);
  float m0 = fmaxf(v.x + bb.x, 0.f);
  float m1 = fmaxf(v.y + bb.y, 0.f);
  atomicAdd(agg + d * 64 + lane * 2 + 0, m0);
  atomicAdd(agg + d * 64 + lane * 2 + 1, m1);
}

// ---------------- stage-2 cross edges: agg[i+HALF] += relu(h[i] + basec[i/NN]) ----------------
__global__ __launch_bounds__(256) void cross_kernel(const float* __restrict__ h,
                                                    const float* __restrict__ basec,
                                                    float* __restrict__ agg) {
  unsigned int gw = (blockIdx.x * 256u + threadIdx.x) >> 5;  // i in [0, HALF_)
  int lane = threadIdx.x & 31;
  if (gw >= (unsigned)HALF_) return;
  unsigned int b = gw >> 6;
  const float2 v  = *(const float2*)(h + (size_t)gw * 64 + lane * 2);
  const float2 bb = *(const float2*)(basec + (size_t)b * 64 + lane * 2);
  float* p = agg + ((size_t)gw + HALF_) * 64 + lane * 2;
  p[0] += fmaxf(v.x + bb.x, 0.f);   // dst unique per i; interior pass already done
  p[1] += fmaxf(v.y + bb.y, 0.f);
}

// ---------------- per-layer edge bias: basec[b] = los_emb[los[b]]@W + b ; base0 from los_emb[0] ----------------
__global__ __launch_bounds__(256) void edgebase_kernel(const int* __restrict__ los,
                                                       const float* __restrict__ los_emb,
                                                       const float* __restrict__ elin_w,
                                                       const float* __restrict__ elin_b,
                                                       int l, float* __restrict__ basec,
                                                       float* __restrict__ base0) {
  unsigned int gw = (blockIdx.x * 256u + threadIdx.x) >> 5;  // [0, B_] ; gw==B_ -> base0
  int lane = threadIdx.x & 31;
  if (gw > (unsigned)B_) return;
  int li = (gw == (unsigned)B_) ? 0 : los[gw];
  const float* e = los_emb + (size_t)li * LD_;
  const float* W = elin_w + (size_t)l * LD_ * 64;
  const float* bb = elin_b + (size_t)l * 64;
  float* outp = (gw == (unsigned)B_) ? base0 : (basec + (size_t)gw * 64);
#pragma unroll
  for (int c2 = 0; c2 < 2; ++c2) {
    int c = lane * 2 + c2;
    float s = bb[c];
#pragma unroll
    for (int k = 0; k < LD_; ++k) s += e[k] * W[k * 64 + c];
    outp[c] = s;
  }
}

// ---------------- GIN node MLP via WMMA: h = relu(LN(((1+eps)h + agg)@W1+b1)) @W2 + b2 ----------------
__global__ __launch_bounds__(256) void mlp_kernel(float* __restrict__ h,
                                                  const float* __restrict__ agg,
                                                  const float* __restrict__ w1,
                                                  const float* __restrict__ b1,
                                                  const float* __restrict__ g,
                                                  const float* __restrict__ be,
                                                  const float* __restrict__ w2,
                                                  const float* __restrict__ b2,
                                                  const float* __restrict__ eps, int l) {
  __shared__ _Float16 sW1[64 * 64];
  __shared__ _Float16 sW2[64 * 64];
  __shared__ float sB1[64], sB2[64], sG[64], sBe[64];
  __shared__ _Float16 sStage[8][16 * 64];   // per-wave 16x64 tile

  int tid = threadIdx.x;
  for (int i = tid; i < 4096; i += 256) {
    sW1[i] = (_Float16)w1[i];
    sW2[i] = (_Float16)w2[i];
  }
  if (tid < 64) { sB1[tid] = b1[tid]; sB2[tid] = b2[tid]; sG[tid] = g[tid]; sBe[tid] = be[tid]; }
  float e1 = 1.0f + eps[l];
  __syncthreads();

  int wave = tid >> 5, lane = tid & 31;
  size_t rowBase = (size_t)blockIdx.x * 128 + (size_t)wave * 16;
  _Float16* stage = sStage[wave];

  // Y = (1+eps)*h + agg  ->  f16 staging tile
  for (int i = lane; i < 256; i += 32) {
    int r = i >> 4, c = (i & 15) << 2;
    size_t off = (rowBase + r) * 64 + c;
    const float4 hv = *(const float4*)(h + off);
    const float4 av = *(const float4*)(agg + off);
    stage[r * 64 + c + 0] = (_Float16)(e1 * hv.x + av.x);
    stage[r * 64 + c + 1] = (_Float16)(e1 * hv.y + av.y);
    stage[r * 64 + c + 2] = (_Float16)(e1 * hv.z + av.z);
    stage[r * 64 + c + 3] = (_Float16)(e1 * hv.w + av.w);
  }
  __syncthreads();

  int m = lane & 15;
  int khalf = (lane >> 4) << 3;

  // GEMM1: 16x64 = A(16x64) @ W1(64x64)
  v16h a0 = load_a_frag(&stage[m * 64], 0, khalf);
  v16h a1 = load_a_frag(&stage[m * 64], 32, khalf);
  v8f acc[4];
#pragma unroll
  for (int n = 0; n < 4; ++n) {
    v8f c = {};
    c = __builtin_amdgcn_wmma_f32_16x16x32_f16(false, a0, false,
        load_b_frag(sW1, 0, n, lane), (short)0, c, false, false);
    c = __builtin_amdgcn_wmma_f32_16x16x32_f16(false, a1, false,
        load_b_frag(sW1, 1, n, lane), (short)0, c, false, false);
    float bias = sB1[n * 16 + m];
#pragma unroll
    for (int v = 0; v < 8; ++v) c[v] += bias;
    acc[n] = c;
  }

  // LayerNorm per row (C/D layout: VGPR v holds row v (+8 for upper half), cols stripe 16 lanes)
#pragma unroll
  for (int v = 0; v < 8; ++v) {
    float s = acc[0][v] + acc[1][v] + acc[2][v] + acc[3][v];
    float q = acc[0][v] * acc[0][v] + acc[1][v] * acc[1][v] +
              acc[2][v] * acc[2][v] + acc[3][v] * acc[3][v];
#pragma unroll
    for (int msk = 1; msk < 16; msk <<= 1) {
      s += __shfl_xor(s, msk, 32);
      q += __shfl_xor(q, msk, 32);
    }
    float mu   = s * (1.0f / 64.0f);
    float var  = q * (1.0f / 64.0f) - mu * mu;
    float rstd = rsqrtf(var + 1e-5f);
    int row = v + ((lane >> 4) << 3);
#pragma unroll
    for (int n = 0; n < 4; ++n) {
      int col = n * 16 + m;
      float z = (acc[n][v] - mu) * rstd * sG[col] + sBe[col];
      stage[row * 64 + col] = (_Float16)fmaxf(z, 0.0f);
    }
  }
  __syncthreads();

  // GEMM2 + store back to h (in place; each wave owns its 16 rows)
  a0 = load_a_frag(&stage[m * 64], 0, khalf);
  a1 = load_a_frag(&stage[m * 64], 32, khalf);
#pragma unroll
  for (int n = 0; n < 4; ++n) {
    v8f c = {};
    c = __builtin_amdgcn_wmma_f32_16x16x32_f16(false, a0, false,
        load_b_frag(sW2, 0, n, lane), (short)0, c, false, false);
    c = __builtin_amdgcn_wmma_f32_16x16x32_f16(false, a1, false,
        load_b_frag(sW2, 1, n, lane), (short)0, c, false, false);
    float bias = sB2[n * 16 + m];
#pragma unroll
    for (int v = 0; v < 8; ++v) {
      int row = v + ((lane >> 4) << 3);
      h[(rowBase + row) * 64 + n * 16 + m] = c[v] + bias;
    }
  }
}

// ---------------- mean-pool over 64 nodes per graph-half ----------------
__global__ __launch_bounds__(256) void pool_kernel(const float* __restrict__ h,
                                                   float* __restrict__ pooled) {
  unsigned int gw = (blockIdx.x * 256u + threadIdx.x) >> 5;  // [0, 2B)
  int lane = threadIdx.x & 31;
  if (gw >= 2u * B_) return;
  const float* p = h + (size_t)gw * 64 * 64 + lane * 2;
  float s0 = 0.f, s1 = 0.f;
#pragma unroll 8
  for (int j = 0; j < 64; ++j) { s0 += p[j * 64]; s1 += p[j * 64 + 1]; }
  pooled[(size_t)gw * 64 + lane * 2 + 0] = s0 * (1.0f / 64.0f);
  pooled[(size_t)gw * 64 + lane * 2 + 1] = s1 * (1.0f / 64.0f);
}

// ---------------- head: projections, score MLP, softmax fuse, classifier ----------------
__global__ __launch_bounds__(256) void head_kernel(const float* __restrict__ pooled,
                                                   const float* __restrict__ pooled2,
                                                   const float* __restrict__ Wad, const float* __restrict__ bad,
                                                   const float* __restrict__ Wdis, const float* __restrict__ bdis,
                                                   const float* __restrict__ Wm, const float* __restrict__ bm,
                                                   const float* __restrict__ sw1, const float* __restrict__ sb1,
                                                   const float* __restrict__ sw2, const float* __restrict__ sb2,
                                                   const float* __restrict__ cw1, const float* __restrict__ cb1,
                                                   const float* __restrict__ cw2, const float* __restrict__ cb2,
                                                   float* __restrict__ out) {
  __shared__ float sCat[8][192];
  __shared__ float sFus[8][64];
  int wave = threadIdx.x >> 5, lane = threadIdx.x & 31;
  int b = blockIdx.x * 8 + wave;                 // grid == B_/8, always valid
  const float* ad = pooled + (size_t)b * 64;
  const float* ds = pooled + ((size_t)B_ + b) * 64;
  const float* m1 = pooled2 + (size_t)b * 64;
  const float* m2 = pooled2 + ((size_t)B_ + b) * 64;

  float Av[2], Bv[2], Cv[2];
#pragma unroll
  for (int c2 = 0; c2 < 2; ++c2) {
    int c = lane * 2 + c2;
    float a = bad[c], d = bdis[c], mm = bm[c];
    for (int k = 0; k < 64; ++k) {
      float mk = 0.5f * (m1[k] + m2[k]);
      a  += ad[k] * Wad[k * 64 + c];
      d  += ds[k] * Wdis[k * 64 + c];
      mm += mk * Wm[k * 64 + c];
    }
    Av[c2] = a; Bv[c2] = d; Cv[c2] = mm;
    sCat[wave][c] = a; sCat[wave][64 + c] = d; sCat[wave][128 + c] = mm;
  }
  __syncthreads();

  // score layer 1 (192 -> 192), lane owns 6 outputs
  float hid[6];
#pragma unroll
  for (int t = 0; t < 6; ++t) {
    int j = lane * 6 + t;
    float s = sb1[j];
    for (int k = 0; k < 192; ++k) s += sCat[wave][k] * sw1[(size_t)k * 192 + j];
    hid[t] = fmaxf(s, 0.f);
  }
  // score layer 2 (192 -> 3) + wave reduction
  float lg[3];
#pragma unroll
  for (int j = 0; j < 3; ++j) {
    float s = 0.f;
#pragma unroll
    for (int t = 0; t < 6; ++t) s += hid[t] * sw2[(size_t)(lane * 6 + t) * 3 + j];
#pragma unroll
    for (int msk = 1; msk < 32; msk <<= 1) s += __shfl_xor(s, msk, 32);
    lg[j] = s + sb2[j];
  }
  float mx = fmaxf(lg[0], fmaxf(lg[1], lg[2]));
  float e0 = expf(lg[0] - mx), e1x = expf(lg[1] - mx), e2 = expf(lg[2] - mx);
  float inv = 1.0f / (e0 + e1x + e2);
  float w0 = e0 * inv, w1v = e1x * inv, w2v = e2 * inv;
#pragma unroll
  for (int c2 = 0; c2 < 2; ++c2)
    sFus[wave][lane * 2 + c2] = w0 * Av[c2] + w1v * Bv[c2] + w2v * Cv[c2];
  __syncthreads();

  // classifier: 64 -> 128 -> 1, lane owns 4 hidden units
  float r = 0.f;
#pragma unroll
  for (int t = 0; t < 4; ++t) {
    int j = lane * 4 + t;
    float s = cb1[j];
    for (int k = 0; k < 64; ++k) s += sFus[wave][k] * cw1[(size_t)k * 128 + j];
    r += fmaxf(s, 0.f) * cw2[j];
  }
#pragma unroll
  for (int msk = 1; msk < 32; msk <<= 1) r += __shfl_xor(r, msk, 32);
  if (lane == 0) out[b] = r + cb2[0];
}

// ---------------- host orchestration ----------------
extern "C" void kernel_launch(void* const* d_in, const int* in_sizes, int n_in,
                              void* d_out, int out_size, void* d_ws, size_t ws_size,
                              hipStream_t stream) {
  const int*   x        = (const int*)d_in[0];
  const int*   los      = (const int*)d_in[1];
  const int*   edge     = (const int*)d_in[2];
  const float* col_emb  = (const float*)d_in[3];
  const float* los_emb  = (const float*)d_in[4];
  const float* g1_w1    = (const float*)d_in[5];
  const float* g1_b1    = (const float*)d_in[6];
  const float* g1_g     = (const float*)d_in[7];
  const float* g1_be    = (const float*)d_in[8];
  const float* g1_w2    = (const float*)d_in[9];
  const float* g1_b2    = (const float*)d_in[10];
  const float* eps1     = (const float*)d_in[11];
  const float* g2_w1    = (const float*)d_in[12];
  const float* g2_b1    = (const float*)d_in[13];
  const float* g2_g     = (const float*)d_in[14];
  const float* g2_be    = (const float*)d_in[15];
  const float* g2_w2    = (const float*)d_in[16];
  const float* g2_b2    = (const float*)d_in[17];
  const float* eps2     = (const float*)d_in[18];
  const float* elin_w   = (const float*)d_in[19];
  const float* elin_b   = (const float*)d_in[20];
  const float* proj_ad_w  = (const float*)d_in[21];
  const float* proj_ad_b  = (const float*)d_in[22];
  const float* proj_dis_w = (const float*)d_in[23];
  const float* proj_dis_b = (const float*)d_in[24];
  const float* proj_m_w   = (const float*)d_in[25];
  const float* proj_m_b   = (const float*)d_in[26];
  const float* score_w1 = (const float*)d_in[27];
  const float* score_b1 = (const float*)d_in[28];
  const float* score_w2 = (const float*)d_in[29];
  const float* score_b2 = (const float*)d_in[30];
  const float* clf_w1   = (const float*)d_in[31];
  const float* clf_b1   = (const float*)d_in[32];
  const float* clf_w2   = (const float*)d_in[33];
  const float* clf_b2   = (const float*)d_in[34];

  float* h       = (float*)d_ws;                       // N_*64
  float* agg     = h + (size_t)N_ * 64;                // N_*64
  float* pooled  = agg + (size_t)N_ * 64;              // 2B*64
  float* pooled2 = pooled + (size_t)2 * B_ * 64;       // 2B*64
  float* basec   = pooled2 + (size_t)2 * B_ * 64;      // B*64
  float* base0   = basec + (size_t)B_ * 64;            // 64

  const int* src = edge;
  const int* dst = edge + E_;

  const unsigned int aggN4   = (unsigned int)((size_t)N_ * 64 / 4);
  const unsigned int zeroGrd = (aggN4 + 255u) / 256u;

  embed_kernel<<<(N_ * 32) / 256, 256, 0, stream>>>(x, col_emb, h);

  // ---- stage 1: three GIN layers ----
  for (int l = 0; l < 3; ++l) {
    int i = (l < 1) ? 0 : 1;
    zero_kernel<<<zeroGrd, 256, 0, stream>>>((float4*)agg, aggN4);
    scatter1_kernel<<<E_ / 8, 256, 0, stream>>>(h, src, dst, agg);
    mlp_kernel<<<N_ / 128, 256, 0, stream>>>(h, agg,
        g1_w1 + (size_t)i * 4096, g1_b1 + i * 64, g1_g + i * 64, g1_be + i * 64,
        g1_w2 + (size_t)i * 4096, g1_b2 + i * 64, eps1, l);
  }
  pool_kernel<<<(2 * B_) / 8, 256, 0, stream>>>(h, pooled);

  // ---- stage 2: three edge-feature GIN layers ----
  for (int l = 0; l < 3; ++l) {
    int i = (l < 1) ? 0 : 1;
    zero_kernel<<<zeroGrd, 256, 0, stream>>>((float4*)agg, aggN4);
    edgebase_kernel<<<(B_ + 1 + 7) / 8, 256, 0, stream>>>(los, los_emb, elin_w, elin_b,
                                                          l, basec, base0);
    scatter2_kernel<<<E_ / 8, 256, 0, stream>>>(h, src, dst, base0, agg);
    cross_kernel<<<HALF_ / 8, 256, 0, stream>>>(h, basec, agg);
    mlp_kernel<<<N_ / 128, 256, 0, stream>>>(h, agg,
        g2_w1 + (size_t)i * 4096, g2_b1 + i * 64, g2_g + i * 64, g2_be + i * 64,
        g2_w2 + (size_t)i * 4096, g2_b2 + i * 64, eps2, l);
  }
  pool_kernel<<<(2 * B_) / 8, 256, 0, stream>>>(h, pooled2);

  head_kernel<<<B_ / 8, 256, 0, stream>>>(pooled, pooled2,
      proj_ad_w, proj_ad_b, proj_dis_w, proj_dis_b, proj_m_w, proj_m_b,
      score_w1, score_b1, score_w2, score_b2,
      clf_w1, clf_b1, clf_w2, clf_b2, (float*)d_out);
}
